// RotaryMultiheadAttention_80169859547320
// MI455X (gfx1250) — matrix operand order repaired and checked by reading, not compiled, not run on hardware
//
#include <hip/hip_runtime.h>
#include <cstdint>

// ---------------------------------------------------------------------------
// RotaryMultiheadAttention forward for MI455X (gfx1250, wave32, WMMA).
// All GEMMs run on v_wmma_f32_16x16x32_bf16 (fp32 accumulate).
//
// ws layout (bytes, 256B aligned chunks), total ~208 MiB:
//   xb   bf16 [4096][2048]          x converted
//   wqb/wkb/wvb/wob bf16 [2048][2048]
//   Qf/Kf/Vf f32 [4096][2048]       projection outputs (token-major)
//   Qh/Kh bf16 [32][2048][128]      head-major, RoPE applied (Q pre-scaled)
//   Vt   bf16 [32][128][2048]       head-major, dh-major (transposed for PV)
//   Ab   bf16 [4096][2048]          attention output, token-major
// ---------------------------------------------------------------------------

typedef __attribute__((ext_vector_type(16))) __bf16        v16bf;
typedef __attribute__((ext_vector_type(8)))  float         v8f;
typedef __attribute__((ext_vector_type(4)))  unsigned int  v4u;

union Frag16 { v16bf h; v4u q[2]; };

__device__ __forceinline__ unsigned short f2bf(float x) {
  unsigned int u = __float_as_uint(x);
  u += 0x7FFFu + ((u >> 16) & 1u);          // round-to-nearest-even
  return (unsigned short)(u >> 16);
}

// A-matrix fragment (16xK tile, 16-bit elems), ISA 7.12.2:
//   lane<16 : row M=lane,   K = {kk+0..7,  kk+16..23}
//   lane>=16: row M=lane-16,K = {kk+8..15, kk+24..31}
// base points at row 0 of the tile.
__device__ __forceinline__ Frag16 load_a_frag(const unsigned short* base, int ld,
                                              int kk, int lane) {
  const int half = lane >> 4, lidx = lane & 15;
  const unsigned short* p = base + (size_t)lidx * ld + kk + half * 8;
  Frag16 f;
  f.q[0] = *reinterpret_cast<const v4u*>(p);
  f.q[1] = *reinterpret_cast<const v4u*>(p + 16);
  return f;
}

// B-matrix fragment (Kx16 tile, 16-bit elems):
//   lane<16 : col N=lane,    K = {kk+0..7,  kk+16..23}
//   lane>=16: col N=lane-16, K = {kk+8..15, kk+24..31}
// base points at "column row 0": memory row n holds B[:,n] contiguously in K.
__device__ __forceinline__ Frag16 load_b_frag(const unsigned short* base, int ld,
                                              int kk, int lane) {
  const int half = lane >> 4, lidx = lane & 15;
  const unsigned short* p = base + (size_t)lidx * ld + kk + half * 8;
  Frag16 f;
  f.q[0] = *reinterpret_cast<const v4u*>(p);
  f.q[1] = *reinterpret_cast<const v4u*>(p + 16);
  return f;
}

__device__ __forceinline__ v8f wmma_bf16(const Frag16& a, const Frag16& b, v8f c) {
  return __builtin_amdgcn_wmma_f32_16x16x32_bf16(false, a.h, false, b.h,
                                                 (short)0, c, false, false);
}

// ---------------------------------------------------------------------------
// fp32 -> bf16 conversion, 4 elements/thread
// ---------------------------------------------------------------------------
__global__ __launch_bounds__(256) void cvt_f32_bf16(
    const float* __restrict__ in, unsigned short* __restrict__ out, int n4) {
  int i = blockIdx.x * 256 + threadIdx.x;
  if (i >= n4) return;
  float4 v = reinterpret_cast<const float4*>(in)[i];
  uint2 o;
  o.x = (unsigned)f2bf(v.x) | ((unsigned)f2bf(v.y) << 16);
  o.y = (unsigned)f2bf(v.z) | ((unsigned)f2bf(v.w) << 16);
  reinterpret_cast<uint2*>(out)[i] = o;
}

// ---------------------------------------------------------------------------
// C[M,N] = A[M,K] * W[N,K]^T + bias, N = K = 2048 (compile-time).
// One wave -> 32x64 tile (2 A-frags x 4 B-frags, 8 accumulators).
// Software-pipelined: next K-step's 12 b128 loads are issued before the
// current K-step's 8 WMMAs, so s_wait_loadcnt lands after the WMMA burst.
// block = 128 (4 waves), grid = (M/32, 2048/256)
// ---------------------------------------------------------------------------
__global__ __launch_bounds__(128) void gemm_bf16_nt(
    const unsigned short* __restrict__ A,
    const unsigned short* __restrict__ W,
    const float* __restrict__ bias,
    float* __restrict__ C) {
  constexpr int N = 2048, K = 2048;
  const int lane = threadIdx.x & 31;
  const int wave = threadIdx.x >> 5;
  const int half = lane >> 4, lidx = lane & 15;
  const int tileM = blockIdx.x * 32;
  const int tileN = (blockIdx.y * 4 + wave) * 64;

  const unsigned short* a0 = A + (size_t)tileM * K;
  const unsigned short* a1 = A + (size_t)(tileM + 16) * K;
  const unsigned short* wb = W + (size_t)tileN * K;

  v8f acc[2][4] = {};
  Frag16 af[2], bf[4];

  af[0] = load_a_frag(a0, K, 0, lane);
  af[1] = load_a_frag(a1, K, 0, lane);
#pragma unroll
  for (int j = 0; j < 4; ++j) bf[j] = load_b_frag(wb + (size_t)(j * 16) * K, K, 0, lane);

  for (int kk = 32; kk < K; kk += 32) {
    // prefetch two K-steps ahead (one 128B line covers a 64-elem K span)
    int kp = (kk + 64 < K) ? kk + 64 : kk;
    __builtin_prefetch(a0 + (size_t)lane * K + kp, 0, 0);
    __builtin_prefetch(wb + (size_t)lane * K + kp, 0, 0);
    __builtin_prefetch(wb + (size_t)(lane + 32) * K + kp, 0, 0);

    Frag16 afn[2], bfn[4];
    afn[0] = load_a_frag(a0, K, kk, lane);
    afn[1] = load_a_frag(a1, K, kk, lane);
#pragma unroll
    for (int j = 0; j < 4; ++j) bfn[j] = load_b_frag(wb + (size_t)(j * 16) * K, K, kk, lane);

#pragma unroll
    for (int j = 0; j < 4; ++j) {
      acc[0][j] = wmma_bf16(af[0], bf[j], acc[0][j]);
      acc[1][j] = wmma_bf16(af[1], bf[j], acc[1][j]);
    }
    af[0] = afn[0]; af[1] = afn[1];
#pragma unroll
    for (int j = 0; j < 4; ++j) bf[j] = bfn[j];
  }
  // drain
#pragma unroll
  for (int j = 0; j < 4; ++j) {
    acc[0][j] = wmma_bf16(af[0], bf[j], acc[0][j]);
    acc[1][j] = wmma_bf16(af[1], bf[j], acc[1][j]);
  }

  // C/D layout: VGPR r, lane = n + 16*(m/8), m = r + 8*half
#pragma unroll
  for (int j = 0; j < 4; ++j) {
    float bv = bias[tileN + j * 16 + lidx];
#pragma unroll
    for (int mb = 0; mb < 2; ++mb)
#pragma unroll
      for (int r = 0; r < 8; ++r) {
        int row = tileM + mb * 16 + half * 8 + r;
        C[(size_t)row * N + tileN + j * 16 + lidx] = acc[mb][j][r] + bv;
      }
  }
}

// ---------------------------------------------------------------------------
// RoPE + transpose to head-major bf16.  One thread = (bh, s, j), j in [0,64),
// handling the rotation pair (j, j+64).  Q gets the 1/sqrt(Dh) scale folded in.
// ---------------------------------------------------------------------------
__global__ __launch_bounds__(256) void rope_transpose(
    const float* __restrict__ Qf, const float* __restrict__ Kf,
    const float* __restrict__ Vf,
    unsigned short* __restrict__ Qh, unsigned short* __restrict__ Kh,
    unsigned short* __restrict__ Vt) {
  constexpr int S = 2048, D = 2048, Dh = 128;
  int idx = blockIdx.x * 256 + threadIdx.x;        // < 32*2048*64
  int j  = idx & 63;
  int s  = (idx >> 6) & (S - 1);
  int bh = idx >> 17;
  int b = bh >> 4, h = bh & 15;
  int tok = b * S + s;

  float inv_freq = __powf(10000.0f, -(float)(2 * j) * (1.0f / 128.0f));
  float sn, cs;
  __sincosf((float)s * inv_freq, &sn, &cs);
  const float scale = 0.08838834764831845f;        // 128^-0.5

  size_t tm = (size_t)tok * D + h * Dh;
  float q0 = Qf[tm + j], q1 = Qf[tm + j + 64];
  float k0 = Kf[tm + j], k1 = Kf[tm + j + 64];
  float v0 = Vf[tm + j], v1 = Vf[tm + j + 64];

  size_t hm = ((size_t)bh * S + s) * Dh;
  Qh[hm + j]      = f2bf((q0 * cs - q1 * sn) * scale);
  Qh[hm + j + 64] = f2bf((q1 * cs + q0 * sn) * scale);
  Kh[hm + j]      = f2bf(k0 * cs - k1 * sn);
  Kh[hm + j + 64] = f2bf(k1 * cs + k0 * sn);
  Vt[((size_t)bh * Dh + j)      * S + s] = f2bf(v0);
  Vt[((size_t)bh * Dh + j + 64) * S + s] = f2bf(v1);
}

// ---------------------------------------------------------------------------
// Flash attention: one wave per (bh, 16-query tile).  Online softmax,
// 32 keys per iteration: 8 score WMMAs + 8 PV WMMAs.
// ---------------------------------------------------------------------------
__global__ __launch_bounds__(32) void flash_attn(
    const unsigned short* __restrict__ Qh, const unsigned short* __restrict__ Kh,
    const unsigned short* __restrict__ Vt, unsigned short* __restrict__ Ab) {
  constexpr int S = 2048, D = 2048, Dh = 128;
  const int qt = blockIdx.x;              // query tile (16 rows)
  const int bh = blockIdx.y;              // b*16 + h
  const int b = bh >> 4, h = bh & 15;
  const int lane = threadIdx.x;
  const int half = lane >> 4, lidx = lane & 15;

  __shared__ alignas(16) unsigned short pbuf[16 * 32];   // P tile staging

  const unsigned short* Qbase = Qh + ((size_t)bh * S + qt * 16) * Dh;
  const unsigned short* Kbase = Kh + (size_t)bh * S * Dh;
  const unsigned short* Vbase = Vt + (size_t)bh * Dh * S;

  Frag16 qf[4];
#pragma unroll
  for (int t = 0; t < 4; ++t) qf[t] = load_a_frag(Qbase, Dh, t * 32, lane);

  v8f o[8] = {};
  float m_i[8], l_i[8];
#pragma unroll
  for (int r = 0; r < 8; ++r) { m_i[r] = -3.0e38f; l_i[r] = 0.0f; }

  for (int key0 = 0; key0 < S; key0 += 32) {
    // warm the next 32-key K tile while this iteration's softmax runs
    int kp = (key0 + 32 < S) ? key0 + 32 : key0;
    __builtin_prefetch(Kbase + (size_t)(kp + lane) * Dh, 0, 0);

    // ---- scores: 16 queries x 32 keys, contraction over Dh=128 ----
    v8f s0 = {}, s1 = {};
#pragma unroll
    for (int t = 0; t < 4; ++t) {
      Frag16 k0 = load_b_frag(Kbase + (size_t)key0 * Dh,        Dh, t * 32, lane);
      Frag16 k1 = load_b_frag(Kbase + (size_t)(key0 + 16) * Dh, Dh, t * 32, lane);
      s0 = wmma_bf16(qf[t], k0, s0);
      s1 = wmma_bf16(qf[t], k1, s1);
    }

    // ---- online softmax (rows live across the 16 lanes of each half) ----
    float corr[8], rsum[8];
#pragma unroll
    for (int r = 0; r < 8; ++r) {
      float mx = fmaxf(s0[r], s1[r]);
#pragma unroll
      for (int msk = 1; msk < 16; msk <<= 1) mx = fmaxf(mx, __shfl_xor(mx, msk));
      float nm = fmaxf(m_i[r], mx);
      corr[r] = __expf(m_i[r] - nm);
      m_i[r] = nm;
      float p0 = __expf(s0[r] - nm);
      float p1 = __expf(s1[r] - nm);
      s0[r] = p0; s1[r] = p1;
      float sm = p0 + p1;
#pragma unroll
      for (int msk = 1; msk < 16; msk <<= 1) sm += __shfl_xor(sm, msk);
      rsum[r] = sm;
    }
#pragma unroll
    for (int r = 0; r < 8; ++r) l_i[r] = l_i[r] * corr[r] + rsum[r];
#pragma unroll
    for (int f = 0; f < 8; ++f)
#pragma unroll
      for (int r = 0; r < 8; ++r) o[f][r] *= corr[r];

    // ---- re-lay P (C layout) -> A layout via LDS ----
#pragma unroll
    for (int r = 0; r < 8; ++r) {
      int m = half * 8 + r;
      pbuf[m * 32 + lidx]      = f2bf(s0[r]);
      pbuf[m * 32 + 16 + lidx] = f2bf(s1[r]);
    }
    __syncthreads();   // single wave -> cheap; orders ds store vs ds load
    Frag16 pf;
    {
      const unsigned short* p = &pbuf[lidx * 32 + half * 8];
      pf.q[0] = *reinterpret_cast<const v4u*>(p);
      pf.q[1] = *reinterpret_cast<const v4u*>(p + 16);
    }
    __syncthreads();

    // ---- PV: contraction over the 32 keys, N = Dh in 8 tiles ----
#pragma unroll
    for (int f = 0; f < 8; ++f) {
      Frag16 vf = load_b_frag(Vbase + (size_t)(f * 16) * S, S, key0, lane);
      o[f] = wmma_bf16(pf, vf, o[f]);
    }
  }

  // ---- epilogue: normalize, store bf16 token-major ----
  float inv_l[8];
#pragma unroll
  for (int r = 0; r < 8; ++r) inv_l[r] = 1.0f / l_i[r];
#pragma unroll
  for (int f = 0; f < 8; ++f)
#pragma unroll
    for (int r = 0; r < 8; ++r) {
      int m = half * 8 + r;
      size_t tok = (size_t)b * S + qt * 16 + m;
      Ab[tok * D + h * Dh + f * 16 + lidx] = f2bf(o[f][r] * inv_l[r]);
    }
}

// ---------------------------------------------------------------------------
extern "C" void kernel_launch(void* const* d_in, const int* in_sizes, int n_in,
                              void* d_out, int out_size, void* d_ws, size_t ws_size,
                              hipStream_t stream) {
  (void)in_sizes; (void)n_in; (void)out_size; (void)ws_size;
  const float* x     = (const float*)d_in[0];
  const float* q_w   = (const float*)d_in[1];
  const float* k_w   = (const float*)d_in[2];
  const float* v_w   = (const float*)d_in[3];
  const float* q_b   = (const float*)d_in[4];
  const float* k_b   = (const float*)d_in[5];
  const float* v_b   = (const float*)d_in[6];
  const float* out_w = (const float*)d_in[7];
  const float* out_b = (const float*)d_in[8];

  constexpr int B = 2, S = 2048, D = 2048, H = 16;
  constexpr int T = B * S;                       // 4096 tokens

  char* w = (char*)d_ws;
  size_t off = 0;
  auto alloc = [&](size_t bytes) -> void* {
    void* p = w + off; off += (bytes + 255) & ~(size_t)255; return p;
  };
  unsigned short* xb  = (unsigned short*)alloc((size_t)T * D * 2);
  unsigned short* wqb = (unsigned short*)alloc((size_t)D * D * 2);
  unsigned short* wkb = (unsigned short*)alloc((size_t)D * D * 2);
  unsigned short* wvb = (unsigned short*)alloc((size_t)D * D * 2);
  unsigned short* wob = (unsigned short*)alloc((size_t)D * D * 2);
  float* Qf = (float*)alloc((size_t)T * D * 4);
  float* Kf = (float*)alloc((size_t)T * D * 4);
  float* Vf = (float*)alloc((size_t)T * D * 4);
  unsigned short* Qhm = (unsigned short*)alloc((size_t)T * D * 2);
  unsigned short* Khm = (unsigned short*)alloc((size_t)T * D * 2);
  unsigned short* Vtr = (unsigned short*)alloc((size_t)T * D * 2);
  unsigned short* Ab  = (unsigned short*)alloc((size_t)T * D * 2);

  // 1) convert inputs to bf16
  {
    int n4 = T * D / 4;
    cvt_f32_bf16<<<(n4 + 255) / 256, 256, 0, stream>>>(x, xb, n4);
    n4 = D * D / 4;
    cvt_f32_bf16<<<(n4 + 255) / 256, 256, 0, stream>>>(q_w,   wqb, n4);
    cvt_f32_bf16<<<(n4 + 255) / 256, 256, 0, stream>>>(k_w,   wkb, n4);
    cvt_f32_bf16<<<(n4 + 255) / 256, 256, 0, stream>>>(v_w,   wvb, n4);
    cvt_f32_bf16<<<(n4 + 255) / 256, 256, 0, stream>>>(out_w, wob, n4);
  }

  // 2) Q/K/V projections
  dim3 ggrid(T / 32, D / 256);
  gemm_bf16_nt<<<ggrid, 128, 0, stream>>>(xb, wqb, q_b, Qf);
  gemm_bf16_nt<<<ggrid, 128, 0, stream>>>(xb, wkb, k_b, Kf);
  gemm_bf16_nt<<<ggrid, 128, 0, stream>>>(xb, wvb, v_b, Vf);

  // 3) RoPE + layout transform
  {
    int nthr = B * H * S * 64;
    rope_transpose<<<nthr / 256, 256, 0, stream>>>(Qf, Kf, Vf, Qhm, Khm, Vtr);
  }

  // 4) attention
  flash_attn<<<dim3(S / 16, B * H), 32, 0, stream>>>(Qhm, Khm, Vtr, Ab);

  // 5) output projection -> d_out (fp32)
  gemm_bf16_nt<<<ggrid, 128, 0, stream>>>(Ab, wob, out_b, (float*)d_out);
}
